// MultiHeadAttention_15496242004641
// MI455X (gfx1250) — matrix operand order repaired and checked by reading, not compiled, not run on hardware
//
#include <hip/hip_runtime.h>

// ---------------------------------------------------------------------------
// MHA for MI455X (gfx1250): bf16 WMMA (v_wmma_f32_16x16x32_bf16) everywhere.
// Pipeline: cvt(fp32->bf16) -> QKV projections (bf16 GEMM, head layouts)
//           -> per-wave flash attention (online softmax, 32 queries/wave,
//              64-key chunks, base-2 domain) -> O projection.
// ---------------------------------------------------------------------------

typedef __attribute__((ext_vector_type(16))) __bf16 v16bf;
typedef __attribute__((ext_vector_type(8)))  __bf16 v8bf;
typedef __attribute__((ext_vector_type(4)))  __bf16 v4bf;
typedef __attribute__((ext_vector_type(8)))  float  v8f;
typedef __attribute__((ext_vector_type(4)))  float  v4f;

#define BATCH 2
#define TT    4096
#define HEADS 8
#define HD    64
#define DMODEL 512
#define NTOK  (BATCH * TT)     // 8192

__device__ __forceinline__ __bf16 f2bf(float f) {
  unsigned u = __builtin_bit_cast(unsigned, f);
  unsigned r = (u + 0x7FFFu + ((u >> 16) & 1u)) >> 16;
  unsigned short s = (unsigned short)r;
  return __builtin_bit_cast(__bf16, s);
}

__device__ __forceinline__ v8f zero8() {
  v8f z;
#pragma unroll
  for (int i = 0; i < 8; ++i) z[i] = 0.0f;
  return z;
}

// A-matrix fragment (16x32, 16-bit): lane half g holds k = 8g..8g+7 (VGPR0-3)
// and 16+8g..16+8g+7 (VGPR4-7). Caller passes  p = row_base + 8*g.
__device__ __forceinline__ v16bf ld_frag_a(const __bf16* p) {
  v8bf lo = *(const v8bf*)(p);
  v8bf hi = *(const v8bf*)(p + 16);
  return __builtin_shufflevector(lo, hi, 0,1,2,3,4,5,6,7,8,9,10,11,12,13,14,15);
}

// B-matrix fragment (32x16, 16-bit): lane half g holds k = 8g..8g+7
// and 16+8g..16+8g+7. Caller passes  p = col_base + 8*g.
__device__ __forceinline__ v16bf ld_frag_b(const __bf16* p) {
  v8bf lo = *(const v8bf*)(p);
  v8bf hi = *(const v8bf*)(p + 16);
  return __builtin_shufflevector(lo, hi, 0,1,2,3,4,5,6,7,8,9,10,11,12,13,14,15);
}

__device__ __forceinline__ v8f wmma_bf16(v16bf a, v16bf b, v8f c) {
  return __builtin_amdgcn_wmma_f32_16x16x32_bf16(
      /*neg_a=*/false, a, /*neg_b=*/false, b,
      /*c_mod=*/(short)0, c, /*reuse_a=*/false, /*reuse_b=*/false);
}

__device__ __forceinline__ float rowred_max16(float x) {
#pragma unroll
  for (int m = 1; m < 16; m <<= 1) x = fmaxf(x, __shfl_xor(x, m, 32));
  return x;
}
__device__ __forceinline__ float rowred_sum16(float x) {
#pragma unroll
  for (int m = 1; m < 16; m <<= 1) x += __shfl_xor(x, m, 32);
  return x;
}

// ---------------------------------------------------------------------------
// fp32 -> bf16 elementwise conversion (vectorized: 4 elems / thread)
// ---------------------------------------------------------------------------
__global__ void cvt_f32_bf16(const float* __restrict__ src,
                             __bf16* __restrict__ dst, int n4) {
  int i = blockIdx.x * 256 + threadIdx.x;
  if (i < n4) {
    v4f f = *(const v4f*)(src + 4 * i);
    v4bf o;
#pragma unroll
    for (int j = 0; j < 4; ++j) o[j] = f2bf(f[j]);
    *(v4bf*)(dst + 4 * i) = o;
  }
}

// ---------------------------------------------------------------------------
// GEMM:  C[m, n] = sum_k X[m, k] * W[n, k] + bias[n]     (torch Linear)
// M = 8192, N = K = 512.  One wave computes a 32(M) x 64(N) tile:
// 2 A-frags + 4 B-frags -> 8 independent WMMAs per 32-wide k-step.
// mode 0/1: write bf16 head-major [B,H,T,Hd]
// mode 2  : write bf16 head-transposed [B,H,Hd,T]   (for V; packed b128)
// mode 3  : write fp32 [M, 512] to d_out            (for O)
// ---------------------------------------------------------------------------
__global__ __launch_bounds__(128) void gemm_bf16(
    const __bf16* __restrict__ X, const __bf16* __restrict__ W,
    const float* __restrict__ bias, __bf16* __restrict__ dstB,
    float* __restrict__ dstF, int mode) {
  const int tid  = threadIdx.x;
  const int wid  = tid >> 5;
  const int lane = tid & 31;
  const int ln16 = lane & 15;
  const int g    = lane >> 4;

  const int m0    = blockIdx.x * 32;
  const int nbase = blockIdx.y * 256 + wid * 64;

  v8f acc[2][4];
#pragma unroll
  for (int q = 0; q < 2; ++q)
#pragma unroll
    for (int nt = 0; nt < 4; ++nt) acc[q][nt] = zero8();

  const __bf16* xrow0 = X + (size_t)(m0 + ln16) * DMODEL + 8 * g;
  const __bf16* xrow1 = X + (size_t)(m0 + 16 + ln16) * DMODEL + 8 * g;
#pragma unroll 4
  for (int kk = 0; kk < DMODEL; kk += 32) {
    v16bf a0 = ld_frag_a(xrow0 + kk);
    v16bf a1 = ld_frag_a(xrow1 + kk);
#pragma unroll
    for (int nt = 0; nt < 4; ++nt) {
      const __bf16* wcol = W + (size_t)(nbase + nt * 16 + ln16) * DMODEL + kk + 8 * g;
      v16bf b = ld_frag_b(wcol);
      acc[0][nt] = wmma_bf16(a0, b, acc[0][nt]);
      acc[1][nt] = wmma_bf16(a1, b, acc[1][nt]);
    }
  }

#pragma unroll
  for (int q = 0; q < 2; ++q) {
#pragma unroll
    for (int nt = 0; nt < 4; ++nt) {
      const int ncol = nbase + nt * 16 + ln16;
      const float bv = bias[ncol];
      const int mrow0 = m0 + 16 * q + 8 * g;         // 8 consecutive rows
      if (mode == 3) {
#pragma unroll
        for (int j = 0; j < 8; ++j)
          dstF[(size_t)(mrow0 + j) * DMODEL + ncol] = acc[q][nt][j] + bv;
      } else if (mode == 2) {
        // [B,H,Hd,T]: 8 consecutive t per lane -> one packed b128 store
        const int b = mrow0 >> 12;
        const int t0 = mrow0 & (TT - 1);
        const int h = ncol >> 6;
        const int hd = ncol & (HD - 1);
        v8bf pk;
#pragma unroll
        for (int j = 0; j < 8; ++j) pk[j] = f2bf(acc[q][nt][j] + bv);
        *(v8bf*)(dstB + (((size_t)(b * HEADS + h) * HD) + hd) * TT + t0) = pk;
      } else {
        const int h = ncol >> 6;
        const int hd = ncol & (HD - 1);
#pragma unroll
        for (int j = 0; j < 8; ++j) {
          const int mrow = mrow0 + j;
          const int b = mrow >> 12;
          const int t = mrow & (TT - 1);
          dstB[(((size_t)(b * HEADS + h) * TT) + t) * HD + hd] = f2bf(acc[q][nt][j] + bv);
        }
      }
    }
  }
}

// ---------------------------------------------------------------------------
// Flash attention: one wave per (b, h, 32-query tile). Causal, online softmax.
// Qh/Kh: [B,H,T,64] bf16;  VhT: [B,H,64,T] bf16;  attnO: [B*T, 512] bf16.
// 64-key chunks: cross-lane softmax reductions amortized over 2x keys;
// causal mask only evaluated in the single chunk that crosses the diagonal;
// softmax runs in base-2 domain (v_exp_f32 direct).  32 WMMAs per chunk.
// ---------------------------------------------------------------------------
__global__ __launch_bounds__(128) void flash_attn(
    const __bf16* __restrict__ Qh, const __bf16* __restrict__ Kh,
    const __bf16* __restrict__ VhT, __bf16* __restrict__ attnO) {
  __shared__ __bf16 pbuf[4][32 * 64];     // per-wave 32 rows x 64 keys

  const int tid  = threadIdx.x;
  const int wid  = tid >> 5;
  const int lane = tid & 31;
  const int ln16 = lane & 15;
  const int g    = lane >> 4;

  const int wave = blockIdx.x * 4 + wid;
  const int bh   = wave >> 7;              // 0..15  (b*8 + h)
  const int qt   = wave & 127;             // 32-query tile index
  const int q0   = qt << 5;

  const __bf16* Qp = Qh  + (size_t)bh * TT * HD;
  const __bf16* Kp = Kh  + (size_t)bh * TT * HD;
  const __bf16* Vp = VhT + (size_t)bh * HD * TT;

  // Q A-fragments: qf[query-subtile][hd-half]
  v16bf qf[2][2];
#pragma unroll
  for (int q = 0; q < 2; ++q) {
    const __bf16* qrow = Qp + (size_t)(q0 + 16 * q + ln16) * HD + 8 * g;
    qf[q][0] = ld_frag_a(qrow);
    qf[q][1] = ld_frag_a(qrow + 32);
  }

  float mrow[2][8], lrow[2][8];
  v8f acc[2][4];
#pragma unroll
  for (int q = 0; q < 2; ++q) {
#pragma unroll
    for (int j = 0; j < 8; ++j) { mrow[q][j] = -__builtin_inff(); lrow[q][j] = 0.0f; }
#pragma unroll
    for (int ht = 0; ht < 4; ++ht) acc[q][ht] = zero8();
  }

  // base-2 domain: exp(s/8 - m) == exp2(s * (log2e/8) - m2)
  const float scale2 = 0.18033688011112042f;   // log2(e) / 8
  const int kend = q0 + 32;                    // keys 0 .. q0+31 (causal)

  for (int kb = 0; kb < kend; kb += 64) {
    // prefetch next chunk's K rows (128B each) and V columns
    if (kb + 64 < kend) {
      __builtin_prefetch(Kp + (size_t)(kb + 64 + lane) * HD, 0, 3);
      __builtin_prefetch(Kp + (size_t)(kb + 96 + lane) * HD, 0, 3);
      __builtin_prefetch(Vp + (size_t)lane * TT + kb + 64, 0, 3);
      __builtin_prefetch(Vp + (size_t)(lane + 32) * TT + kb + 64, 0, 3);
    }

    // ---- K fragments: 4 key subtiles x 2 hd halves (shared by q subtiles) --
    v16bf kf[4][2];
#pragma unroll
    for (int t = 0; t < 4; ++t) {
      const __bf16* kr = Kp + (size_t)(kb + 16 * t + ln16) * HD + 8 * g;
      kf[t][0] = ld_frag_b(kr);
      kf[t][1] = ld_frag_b(kr + 32);
    }

    const bool domask = (kb + 63 > q0);        // only the diagonal chunk

    float alpha[2][8];
#pragma unroll
    for (int q = 0; q < 2; ++q) {
      // ---- scores for this q subtile: 8 WMMAs ----
      v8f s[4];
#pragma unroll
      for (int t = 0; t < 4; ++t) {
        s[t] = wmma_bf16(qf[q][0], kf[t][0], zero8());
        s[t] = wmma_bf16(qf[q][1], kf[t][1], s[t]);
      }

      // ---- online softmax (rows: q0 + 16q + 8g + j) ----
#pragma unroll
      for (int j = 0; j < 8; ++j) {
        float x[4];
#pragma unroll
        for (int t = 0; t < 4; ++t) x[t] = s[t][j] * scale2;
        if (domask) {
          const int qg = q0 + 16 * q + 8 * g + j;
#pragma unroll
          for (int t = 0; t < 4; ++t)
            if (kb + 16 * t + ln16 > qg) x[t] = -__builtin_inff();
        }
        const float cm =
            rowred_max16(fmaxf(fmaxf(x[0], x[1]), fmaxf(x[2], x[3])));
        const float nm = fmaxf(mrow[q][j], cm);
        const float al = __builtin_amdgcn_exp2f(mrow[q][j] - nm);
        float p[4], psum = 0.0f;
#pragma unroll
        for (int t = 0; t < 4; ++t) {
          p[t] = __builtin_amdgcn_exp2f(x[t] - nm);
          psum += p[t];
        }
        const float ps = rowred_sum16(psum);
        lrow[q][j] = lrow[q][j] * al + ps;
        mrow[q][j] = nm;
        alpha[q][j] = al;
        // stage P in C-layout; re-read below in A-layout (same-wave LDS)
        const int prow = (16 * q + 8 * g + j) * 64;
#pragma unroll
        for (int t = 0; t < 4; ++t)
          pbuf[wid][prow + 16 * t + ln16] = f2bf(p[t]);
      }
      // rescale output accumulators (row mapping matches C layout)
#pragma unroll
      for (int ht = 0; ht < 4; ++ht)
#pragma unroll
        for (int j = 0; j < 8; ++j) acc[q][ht][j] *= alpha[q][j];
    }

    // P A-fragments from LDS: 2 q subtiles x 2 key halves
    v16bf pf[2][2];
#pragma unroll
    for (int q = 0; q < 2; ++q) {
      pf[q][0] = ld_frag_a(&pbuf[wid][(16 * q + ln16) * 64 + 8 * g]);
      pf[q][1] = ld_frag_a(&pbuf[wid][(16 * q + ln16) * 64 + 32 + 8 * g]);
    }

    // ---- out += P(16x64) @ V(64k x 64hd): 16 WMMAs over 8 shared V frags ---
#pragma unroll
    for (int ht = 0; ht < 4; ++ht) {
      const __bf16* vrow = Vp + (size_t)(ht * 16 + ln16) * TT + kb + 8 * g;
      v16bf vf0 = ld_frag_b(vrow);
      v16bf vf1 = ld_frag_b(vrow + 32);
      acc[0][ht] = wmma_bf16(pf[0][0], vf0, acc[0][ht]);
      acc[0][ht] = wmma_bf16(pf[0][1], vf1, acc[0][ht]);
      acc[1][ht] = wmma_bf16(pf[1][0], vf0, acc[1][ht]);
      acc[1][ht] = wmma_bf16(pf[1][1], vf1, acc[1][ht]);
    }
  }

  // ---- normalize and write [B*T, 512] bf16 for the O projection ----
  const int b = bh >> 3;
  const int h = bh & 7;
#pragma unroll
  for (int q = 0; q < 2; ++q) {
#pragma unroll
    for (int j = 0; j < 8; ++j) {
      const float inv = 1.0f / lrow[q][j];
      const size_t row =
          (size_t)(b * TT + q0 + 16 * q + 8 * g + j) * DMODEL + h * HD;
#pragma unroll
      for (int ht = 0; ht < 4; ++ht)
        attnO[row + ht * 16 + ln16] = f2bf(acc[q][ht][j] * inv);
    }
  }
}

// ---------------------------------------------------------------------------
extern "C" void kernel_launch(void* const* d_in, const int* in_sizes, int n_in,
                              void* d_out, int out_size, void* d_ws, size_t ws_size,
                              hipStream_t stream) {
  const float* q  = (const float*)d_in[0];
  const float* k  = (const float*)d_in[1];
  const float* v  = (const float*)d_in[2];
  const float* Wq = (const float*)d_in[3];
  const float* bq = (const float*)d_in[4];
  const float* Wk = (const float*)d_in[5];
  const float* bk = (const float*)d_in[6];
  const float* Wv = (const float*)d_in[7];
  const float* bv = (const float*)d_in[8];
  const float* Wo = (const float*)d_in[9];
  const float* bo = (const float*)d_in[10];
  float* out = (float*)d_out;

  const size_t NX = (size_t)NTOK * DMODEL;   // 4,194,304
  const size_t NW = (size_t)DMODEL * DMODEL; //   262,144

  char* base = (char*)d_ws;
  __bf16* qb    = (__bf16*)base;               base += NX * 2;
  __bf16* kb_   = (__bf16*)base;               base += NX * 2;
  __bf16* vb    = (__bf16*)base;               base += NX * 2;
  __bf16* Wqb   = (__bf16*)base;               base += NW * 2;
  __bf16* Wkb   = (__bf16*)base;               base += NW * 2;
  __bf16* Wvb   = (__bf16*)base;               base += NW * 2;
  __bf16* Wob   = (__bf16*)base;               base += NW * 2;
  __bf16* Qhb   = (__bf16*)base;               base += NX * 2;
  __bf16* Khb   = (__bf16*)base;               base += NX * 2;
  __bf16* VhTb  = (__bf16*)base;               base += NX * 2;
  __bf16* attnO = (__bf16*)base;               base += NX * 2;

  // fp32 -> bf16 (4 elems/thread)
  cvt_f32_bf16<<<(int)(NX / 1024), 256, 0, stream>>>(q, qb, (int)(NX / 4));
  cvt_f32_bf16<<<(int)(NX / 1024), 256, 0, stream>>>(k, kb_, (int)(NX / 4));
  cvt_f32_bf16<<<(int)(NX / 1024), 256, 0, stream>>>(v, vb, (int)(NX / 4));
  cvt_f32_bf16<<<(int)(NW / 1024), 256, 0, stream>>>(Wq, Wqb, (int)(NW / 4));
  cvt_f32_bf16<<<(int)(NW / 1024), 256, 0, stream>>>(Wk, Wkb, (int)(NW / 4));
  cvt_f32_bf16<<<(int)(NW / 1024), 256, 0, stream>>>(Wv, Wvb, (int)(NW / 4));
  cvt_f32_bf16<<<(int)(NW / 1024), 256, 0, stream>>>(Wo, Wob, (int)(NW / 4));

  // QKV projections: 32M x 256N per block (4 waves of 32x64)
  dim3 gg(NTOK / 32, DMODEL / 256);   // (256, 2)
  gemm_bf16<<<gg, 128, 0, stream>>>(qb, Wqb, bq, Qhb,  nullptr, 0);
  gemm_bf16<<<gg, 128, 0, stream>>>(kb_, Wkb, bk, Khb,  nullptr, 1);
  gemm_bf16<<<gg, 128, 0, stream>>>(vb, Wvb, bv, VhTb, nullptr, 2);

  // attention: 2*8*128 = 2048 waves, 4 waves/block
  flash_attn<<<512, 128, 0, stream>>>(Qhb, Khb, VhTb, attnO);

  // output projection -> fp32 d_out
  gemm_bf16<<<gg, 128, 0, stream>>>(attnO, Wob, bo, nullptr, out, 3);
}
